// SelfAttn_52192442581087
// MI455X (gfx1250) — matrix operand order repaired and checked by reading, not compiled, not run on hardware
//
#include <hip/hip_runtime.h>
#include <hip/hip_bf16.h>

// Problem constants (fixed by the reference).
#define B_   4
#define C_   128
#define CQK_ 16
#define N_   4096      // H*W = 64*64
#define ES_  4100      // f32 energy row stride in floats  (4100 % 64 == 4 -> conflict-free columns)
#define AS_  4104      // bf16 prob row stride in elements (2052 dwords % 64 == 4 -> conflict-free b128 columns)

typedef __attribute__((ext_vector_type(2)))  float    v2f;
typedef __attribute__((ext_vector_type(4)))  float    v4f;
typedef __attribute__((ext_vector_type(8)))  float    v8f;
typedef __attribute__((ext_vector_type(16))) __bf16   v16bf;
typedef __attribute__((ext_vector_type(4)))  unsigned u4;

union BFPack { v16bf v; unsigned u[8]; u4 q[2]; };

__device__ __forceinline__ unsigned short f2bf(float f) {
  // round-to-nearest-even float -> bf16
  unsigned u = __float_as_uint(f);
  u += 0x7fffu + ((u >> 16) & 1u);
  return (unsigned short)(u >> 16);
}

// ---------------------------------------------------------------------------
// Kernel 1: fused Q/K/V projections with V_WMMA_F32_16X16X4_F32 (exact f32).
//   m-tile 0 -> qT[b][j][16]   (f32, transposed for A-operand loads)
//   m-tile 1 -> kT[b][j][16]   (f32, transposed for B-operand loads)
//   m-tiles 2..9 -> v[b][c][j] (bf16, channel-major: PV B-fragments contiguous)
// ---------------------------------------------------------------------------
__global__ __launch_bounds__(256) void proj_kernel(
    const float* __restrict__ x,
    const float* __restrict__ Wq, const float* __restrict__ bq,
    const float* __restrict__ Wk, const float* __restrict__ bk,
    const float* __restrict__ Wv, const float* __restrict__ bv,
    float* __restrict__ qT, float* __restrict__ kT,
    unsigned short* __restrict__ vbf)
{
  const int tid  = threadIdx.x;
  const int wave = tid >> 5;
  const int lane = tid & 31;
  const int half = lane >> 4;
  const int lm   = lane & 15;

  const int blk = blockIdx.x;
  const int b   = blk / (10 * 32);
  const int rem = blk % (10 * 32);
  const int mt  = rem >> 5;
  const int jg  = rem & 31;
  const int j0  = ((jg << 3) + wave) << 4;

  const float* W;  const float* bias;  int row0;
  if (mt == 0)      { W = Wq; bias = bq; row0 = 0; }
  else if (mt == 1) { W = Wk; bias = bk; row0 = 0; }
  else              { W = Wv; bias = bv; row0 = (mt - 2) << 4; }

  const float* Wrow = W + (size_t)(row0 + lm) * C_;
  const float* xb   = x + (size_t)b * C_ * N_;

  v8f acc = {};
#pragma unroll 8
  for (int kk = 0; kk < 32; ++kk) {
    const int d0 = (kk << 2) + (half << 1);
    v2f a = *(const v2f*)(Wrow + d0);          // A[m][2h],A[m][2h+1]
    v2f bb;                                    // B[2h][n],B[2h+1][n]
    bb.x = xb[(size_t)d0 * N_ + j0 + lm];
    bb.y = xb[(size_t)(d0 + 1) * N_ + j0 + lm];
    acc = __builtin_amdgcn_wmma_f32_16x16x4_f32(false, a, false, bb, (short)0, acc,
                                                false, false);
  }

#pragma unroll
  for (int r = 0; r < 8; ++r) {
    const int m  = r + (half << 3);
    const int j  = j0 + lm;
    const float val = acc[r] + bias[row0 + m];
    if (mt == 0)
      qT[((size_t)b * N_ + j) * CQK_ + m] = val;
    else if (mt == 1)
      kT[((size_t)b * N_ + j) * CQK_ + m] = val;
    else
      vbf[((size_t)b * C_ + row0 + m) * N_ + j] = f2bf(val);
  }
}

// ---------------------------------------------------------------------------
// Kernel 2: fused attention per (batch, 16-query row block). 258 KB LDS/WGP.
//  S1: energy[16][4096] via f32 WMMA -> LDS (never touches HBM)
//  S2: row softmax in LDS; one pass normalizes, writes the 256 MB attention
//      output (nontemporal b128) AND compacts probs to bf16 IN PLACE in a
//      WMMA-fragment-friendly layout (one barrier per row; row c's bf16 bytes
//      [8208c,8208c+8192) never overlap f32 reads [16400c',...) for c'>c).
//  S3: out = attn x V^T: 2 ds_load_b128 + 2 global_load_b128 per bf16 WMMA.
// ---------------------------------------------------------------------------
__global__ __launch_bounds__(256) void attn_kernel(
    const float* __restrict__ qT, const float* __restrict__ kT,
    const unsigned short* __restrict__ vbf,
    const float* __restrict__ x, const float* __restrict__ gamma,
    float* __restrict__ out, float* __restrict__ attn)
{
  extern __shared__ float smem[];
  float*          es  = smem;                         // 16*ES_ f32 (energy -> exp)
  unsigned short* abf = (unsigned short*)smem;        // in-place bf16 probs, stride AS_
  float* qt     = es + 16 * ES_;                      // 256
  float* red    = qt + 256;                           // 256
  float* rowmax = red + 256;                          // 16
  float* rowsum = rowmax + 16;                        // 16

  const int tid  = threadIdx.x;
  const int wave = tid >> 5;
  const int lane = tid & 31;
  const int half = lane >> 4;
  const int lm   = lane & 15;

  const int b  = blockIdx.x >> 8;
  const int i0 = (blockIdx.x & 255) << 4;

  qt[tid] = qT[((size_t)b * N_ + i0 + (tid >> 4)) * CQK_ + (tid & 15)];
  __syncthreads();

  // --- Stage 1: energy = q^T k (exact f32 WMMA) ---
  v2f aq[4];
#pragma unroll
  for (int kk = 0; kk < 4; ++kk) {
    aq[kk].x = qt[lm * 16 + (kk << 2) + (half << 1)];
    aq[kk].y = qt[lm * 16 + (kk << 2) + (half << 1) + 1];
  }
  const float* kTb = kT + (size_t)b * N_ * CQK_;
  for (int jt = wave; jt < 256; jt += 8) {
    const int j0 = jt << 4;
    v8f c = {};
#pragma unroll
    for (int kk = 0; kk < 4; ++kk) {
      v2f bb = *(const v2f*)(kTb + (size_t)(j0 + lm) * CQK_ + (kk << 2) + (half << 1));
      c = __builtin_amdgcn_wmma_f32_16x16x4_f32(false, aq[kk], false, bb, (short)0, c,
                                                false, false);
    }
#pragma unroll
    for (int r = 0; r < 8; ++r)
      es[(r + (half << 3)) * ES_ + j0 + lm] = c[r];
  }
  __syncthreads();

  // --- Stage 2a: row max (b128 scans, lane-distinct bank groups) ---
  const int row = tid >> 4, sub = tid & 15;
  float* erow = es + row * ES_;
  float m_loc = -3.4e38f;
  for (int t = 0; t < 64; ++t) {
    v4f f = *(const v4f*)(erow + (sub << 2) + (t << 6));
    m_loc = fmaxf(m_loc, fmaxf(fmaxf(f.x, f.y), fmaxf(f.z, f.w)));
  }
  red[tid] = m_loc;
  __syncthreads();
  if (sub == 0) {
    float m = red[row * 16];
    for (int u = 1; u < 16; ++u) m = fmaxf(m, red[row * 16 + u]);
    rowmax[row] = m;
  }
  __syncthreads();

  // --- Stage 2b: exp + row sum (p~ written back to es) ---
  const float rmax = rowmax[row];
  float s_loc = 0.f;
  for (int t = 0; t < 64; ++t) {
    v4f f = *(const v4f*)(erow + (sub << 2) + (t << 6));
    f.x = __expf(f.x - rmax); f.y = __expf(f.y - rmax);
    f.z = __expf(f.z - rmax); f.w = __expf(f.w - rmax);
    *(v4f*)(erow + (sub << 2) + (t << 6)) = f;
    s_loc += (f.x + f.y) + (f.z + f.w);
  }
  red[tid] = s_loc;
  __syncthreads();
  if (sub == 0) {
    float s = 0.f;
    for (int u = 0; u < 16; ++u) s += red[row * 16 + u];
    rowsum[row] = s;
  }
  __syncthreads();

  // --- Stage 2c: per row -> normalize, write attention (NT b128), compact to
  //     bf16 in place. Register-stage reads, one barrier per row. ---
  for (int c = 0; c < 16; ++c) {
    const float inv = 1.0f / rowsum[c];
    v4f pv[4];
#pragma unroll
    for (int t = 0; t < 4; ++t)
      pv[t] = *(const v4f*)(es + c * ES_ + (tid << 2) + (t << 10));
    __syncthreads();
    float* arow = attn + ((size_t)b * N_ + i0 + c) * N_;
#pragma unroll
    for (int t = 0; t < 4; ++t) {
      const int j = (tid << 2) + (t << 10);
      v4f p;
      p.x = pv[t].x * inv; p.y = pv[t].y * inv;
      p.z = pv[t].z * inv; p.w = pv[t].w * inv;
      __builtin_nontemporal_store(p, (v4f*)(arow + j));   // 256 MB mandatory write
      unsigned long long w =  (unsigned long long)f2bf(p.x)
                           | ((unsigned long long)f2bf(p.y) << 16)
                           | ((unsigned long long)f2bf(p.z) << 32)
                           | ((unsigned long long)f2bf(p.w) << 48);
      *(unsigned long long*)(abf + c * AS_ + j) = w;
    }
  }
  __syncthreads();

  // --- Stage 3: out(16q x 128c) = attn x V^T, one 16x16 tile per wave.
  //     A: 2x ds_load_b128, B: 2x global_load_b128 per v_wmma_f32_16x16x32_bf16
  const int n0 = wave << 4;
  const unsigned*       vrow   = (const unsigned*)(vbf + ((size_t)b * C_ + n0 + lm) * N_);
  const unsigned short* arow_l = abf + lm * AS_ + (half << 3);
  v8f acc = {};
  for (int kt = 0; kt < 128; ++kt) {
    const int j0 = kt << 5;
    BFPack A, Bv;
    A.q[0]  = *(const u4*)(arow_l + j0);            // A[m][K=2(v&3)+8h+16(v/4)]
    A.q[1]  = *(const u4*)(arow_l + j0 + 16);
    Bv.q[0] = *(const u4*)(vrow + (j0 >> 1) + (half << 2));      // B[K=2(v&3)+8h+16(v/4)][n]
    Bv.q[1] = *(const u4*)(vrow + (j0 >> 1) + (half << 2) + 8);
    acc = __builtin_amdgcn_wmma_f32_16x16x32_bf16(false, A.v, false, Bv.v, (short)0, acc,
                                                  false, false);
  }

  // Epilogue: out = gamma * (attn @ V^T) + x   (8 consecutive f32 per lane)
  const float g = gamma[0];
  const int   c = n0 + lm;
  const float* xr = x   + ((size_t)b * C_ + c) * N_ + i0;
  float*     orow = out + ((size_t)b * C_ + c) * N_ + i0;
#pragma unroll
  for (int r = 0; r < 8; ++r) {
    const int m = r + (half << 3);
    orow[m] = g * acc[r] + xr[m];
  }
}

// ---------------------------------------------------------------------------
extern "C" void kernel_launch(void* const* d_in, const int* in_sizes, int n_in,
                              void* d_out, int out_size, void* d_ws, size_t ws_size,
                              hipStream_t stream) {
  (void)in_sizes; (void)n_in; (void)out_size; (void)ws_size;
  const float* x     = (const float*)d_in[0];
  const float* Wq    = (const float*)d_in[1];
  const float* bq    = (const float*)d_in[2];
  const float* Wk    = (const float*)d_in[3];
  const float* bk    = (const float*)d_in[4];
  const float* Wv    = (const float*)d_in[5];
  const float* bv    = (const float*)d_in[6];
  const float* gamma = (const float*)d_in[7];

  float* out  = (float*)d_out;                       // [B,C,H,W] fp32
  float* attn = out + (size_t)B_ * C_ * N_;          // [B,N,N] fp32

  // Scratch: qT (1 MB) | kT (1 MB) | V bf16 (4 MB)  => 6 MB of d_ws
  float* qT = (float*)d_ws;
  float* kT = qT + (size_t)B_ * N_ * CQK_;
  unsigned short* vbf = (unsigned short*)(kT + (size_t)B_ * N_ * CQK_);

  proj_kernel<<<dim3(B_ * 10 * 32), dim3(256), 0, stream>>>(
      x, Wq, bq, Wk, bk, Wv, bv, qT, kT, vbf);

  const size_t smem = (size_t)(16 * ES_ + 256 + 256 + 32) * sizeof(float); // ~258.4 KB
  (void)hipFuncSetAttribute((const void*)attn_kernel,
                            hipFuncAttributeMaxDynamicSharedMemorySize, (int)smem);
  attn_kernel<<<dim3(B_ * (N_ / 16)), dim3(256), smem, stream>>>(
      qT, kT, vbf, x, gamma, out, attn);
}
